// LSTM_44263932953047
// MI455X (gfx1250) — compile-verified, matrix operand order checked by reading, not run on hardware
//
#include <hip/hip_runtime.h>
#include <stdint.h>

#define T_STEPS 1024
#define BATCH   32
#define IN_DIM  512
#define HID     512
#define G4H     2048          // 4*HID
#define NWG_REC 32            // persistent workgroups in recurrence

typedef __attribute__((ext_vector_type(16))) __bf16 bf16x16;
typedef __attribute__((ext_vector_type(8)))  float  f32x8;
typedef __attribute__((ext_vector_type(4)))  unsigned u32x4;
typedef __attribute__((ext_vector_type(8)))  int      i32x8;
typedef __attribute__((ext_vector_type(4)))  int      i32x4;

union FragAB { unsigned u[8]; bf16x16 v; };

__device__ __forceinline__ unsigned short f2bf(float f) {
  unsigned u = __float_as_uint(f);
  unsigned r = 0x7FFFu + ((u >> 16) & 1u);   // round-to-nearest-even
  return (unsigned short)((u + r) >> 16);
}
// K offsets inside a 32-deep chunk for 16-bit A/B fragments (dword pairs):
// p=0..3 -> k = 0,2,4,6 ; p=4..7 -> k = 16,18,20,22 ; +8 for lanes 16..31
__device__ __forceinline__ int kpat(int p) { return ((p & 3) << 1) | ((p >> 2) << 4); }
__device__ __forceinline__ float sigmoidf(float x) { return 1.0f / (1.0f + __expf(-x)); }

// ---- Tensor Data Mover: 2-D tile of bf16 (2-byte) elements -> LDS ----------
// D# group0: [1:0]count=1, [63:32]=lds_addr, [120:64]=global_addr, [127:126]=2
// D# group1: [17:16]=data_size(1 => 2B), tensor_dim0[79:48], tensor_dim1[111:80],
//            tile_dim0[127:112], tile_dim1[143:128], dim0_stride[207:160]
__device__ __forceinline__ void tdm_load_2d_bf16(unsigned lds_addr,
                                                 const void* gptr,
                                                 unsigned rows,
                                                 unsigned rowlen,
                                                 unsigned stride) {
  unsigned long long ga = (unsigned long long)(uintptr_t)gptr;
  u32x4 g0;
  g0.x = 1u;                                                  // count=1
  g0.y = lds_addr;
  g0.z = (unsigned)(ga & 0xFFFFFFFFu);
  g0.w = (unsigned)((ga >> 32) & 0x01FFFFFFu) | 0x80000000u;  // type=2
  i32x8 g1;
  g1[0] = (int)(1u << 16);                                    // data_size=2B
  g1[1] = (int)((rowlen & 0xFFFFu) << 16);                    // tensor_dim0 lo
  g1[2] = (int)(((rowlen >> 16) & 0xFFFFu) | ((rows & 0xFFFFu) << 16));
  g1[3] = (int)(((rows >> 16) & 0xFFFFu) | ((rowlen & 0xFFFFu) << 16)); // tile_dim0
  g1[4] = (int)(rows & 0xFFFFu);                              // tile_dim1 (tile_dim2=0)
  g1[5] = (int)stride;                                        // dim0_stride lo32
  g1[6] = 0;
  g1[7] = 0;
  i32x4 z4 = {0, 0, 0, 0};
#if __clang_major__ >= 23
  i32x8 z8 = {0, 0, 0, 0, 0, 0, 0, 0};
  __builtin_amdgcn_tensor_load_to_lds(g0, g1, z4, z4, z8, 0);
#else
  __builtin_amdgcn_tensor_load_to_lds(g0, g1, z4, z4, 0);
#endif
}

// ---------------------------------------------------------------- converts
__global__ void cvt_bf16_kernel(const float* __restrict__ src,
                                unsigned short* __restrict__ dst, int n) {
  int i = blockIdx.x * blockDim.x + threadIdx.x;
  if (i < n) dst[i] = f2bf(src[i]);
}

__global__ void init_state_kernel(const float* __restrict__ h0slice,
                                  unsigned short* __restrict__ hbuf,
                                  unsigned* __restrict__ sync) {
  int i = blockIdx.x * blockDim.x + threadIdx.x;
  if (i < BATCH * HID) hbuf[i] = f2bf(h0slice[i]);
  if (i < 2) sync[i] = 0u;
}

// ------------------------------------------------- gi = A @ W^T + b0 + b1
// A: (M,512) fp32 row-major ; W: (2048,512) bf16 row-major ; gi: (M,2048) fp32
// Tile 128(M) x 32(N); B panel (32 rows x full K=512, 32 KB) loaded once by TDM.
__global__ __launch_bounds__(256) void gemm_gates_kernel(
    const float* __restrict__ A, const unsigned short* __restrict__ W,
    const float* __restrict__ bias0, const float* __restrict__ bias1,
    float* __restrict__ gi) {
  __shared__ unsigned short As[128 * 32];   // 8 KB, per-K-chunk A staging
  __shared__ unsigned short Bp[32 * 512];   // 32 KB, full-K weight panel
  const int tid  = threadIdx.x;
  const int lane = tid & 31;
  const int wave = tid >> 5;        // 0..7 : 16-row block each
  const int m0   = blockIdx.y * 128;
  const int n0   = blockIdx.x * 32;
  const int kh   = lane >> 4;
  const int l15  = lane & 15;

  // async-tensor load of the whole B panel (rows n0..n0+31, K=0..511)
  if (wave == 0)
    tdm_load_2d_bf16((unsigned)(uintptr_t)(void*)&Bp[0],
                     W + (size_t)n0 * 512, 32, 512, 512);

  f32x8 acc[2] = {};

  for (int kk = 0; kk < 512; kk += 32) {
    // stage A tile (fp32 -> bf16): 128x32
#pragma unroll
    for (int e = 0; e < 16; ++e) {
      int idx = e * 256 + tid;
      int r = idx >> 5, c = idx & 31;
      As[r * 32 + c] = f2bf(A[(size_t)(m0 + r) * 512 + kk + c]);
    }
    if (kk + 32 < 512)  // hint the next A chunk toward the caches
      __builtin_prefetch(&A[(size_t)(m0 + (tid >> 1)) * 512 + kk + 32], 0, 1);
    if (kk == 0 && wave == 0) __builtin_amdgcn_s_wait_tensorcnt(0);
    __syncthreads();

    FragAB afr, bfr[2];
    {
      int m = wave * 16 + l15;
#pragma unroll
      for (int p = 0; p < 8; ++p)
        afr.u[p] = *(const unsigned*)(&As[m * 32 + kpat(p) + kh * 8]);
    }
#pragma unroll
    for (int tj = 0; tj < 2; ++tj) {
      int n = tj * 16 + l15;
#pragma unroll
      for (int p = 0; p < 8; ++p)
        bfr[tj].u[p] = *(const unsigned*)(&Bp[n * 512 + kk + kpat(p) + kh * 8]);
    }
#pragma unroll
    for (int tj = 0; tj < 2; ++tj)
      acc[tj] = __builtin_amdgcn_wmma_f32_16x16x32_bf16(
          false, afr.v, false, bfr[tj].v, (short)0, acc[tj], false, false);
    __syncthreads();
  }

  // epilogue: D layout -> (m,n); fold both biases
#pragma unroll
  for (int tj = 0; tj < 2; ++tj) {
    int ng = n0 + tj * 16 + l15;
    float b = bias0[ng] + bias1[ng];
#pragma unroll
    for (int r = 0; r < 8; ++r) {
      int mg = m0 + wave * 16 + r + 8 * kh;
      gi[(size_t)mg * G4H + ng] = acc[tj][r] + b;
    }
  }
}

// ------------------------------------------------------ persistent recurrence
// Each WG owns 16 hidden columns (j0..j0+15); its 64 w_hh rows (4 gates x 16)
// are TDM-loaded into LDS (exactly 64 KB). Wave 0 handles batch rows 0..15,
// wave 1 rows 16..31; each wave holds 4 gate accumulator tiles + c tile.
__global__ __launch_bounds__(64) void lstm_rec_kernel(
    const float* __restrict__ gi,            // (T,B,4H) fp32, biases folded
    const unsigned short* __restrict__ whh,  // (2048,512) bf16
    const float* __restrict__ c0,            // (B,H) layer slice
    unsigned short* __restrict__ hbuf,       // 2 x (B*512) bf16 ping-pong
    float* __restrict__ out,                 // (T,B,H) fp32
    float* __restrict__ hn,                  // (B,H)
    float* __restrict__ cn,                  // (B,H)
    unsigned* __restrict__ sync) {           // [0]=counter [1]=generation
  __shared__ unsigned short Ws[64 * 512];    // 64 KB
  const int tid  = threadIdx.x;
  const int lane = tid & 31;
  const int wave = tid >> 5;                 // 0..1
  const int mt   = wave * 16;
  const int j0   = blockIdx.x * 16;
  const int n    = lane & 15;
  const int kh   = lane >> 4;

  // TDM-stage w_hh slice: LDS row rr = q*16 + jloc  <-  row q*512 + j0 + jloc
  if (wave == 0) {
#pragma unroll
    for (int q = 0; q < 4; ++q)
      tdm_load_2d_bf16((unsigned)(uintptr_t)(void*)&Ws[0] +
                           (unsigned)(q * 16 * 512 * 2),
                       whh + (size_t)(q * HID + j0) * 512, 16, 512, 512);
    __builtin_amdgcn_s_wait_tensorcnt(0);
  }
  // persistent c tile in D layout
  float c[8];
#pragma unroll
  for (int r = 0; r < 8; ++r) c[r] = c0[(mt + r + 8 * kh) * HID + j0 + n];
  __syncthreads();

  int cur = 0;
  for (int t = 0; t < T_STEPS; ++t) {
    const unsigned short* hsrc = hbuf + cur * (BATCH * HID);
    f32x8 acc[4];
    // accumulators start at precomputed input projection gi[t]
#pragma unroll
    for (int q = 0; q < 4; ++q)
#pragma unroll
      for (int r = 0; r < 8; ++r)
        acc[q][r] = gi[(size_t)t * BATCH * G4H +
                       (size_t)(mt + r + 8 * kh) * G4H + q * HID + j0 + n];
    if (t + 1 < T_STEPS)  // pull next step's gi toward the caches
      __builtin_prefetch(gi + (size_t)(t + 1) * BATCH * G4H +
                             (size_t)(mt + n) * G4H + j0, 0, 1);

    // gates += h @ w_hh^T  (K = 512, bf16 WMMA, A shared across 4 gates)
    for (int kk = 0; kk < HID; kk += 32) {
      FragAB af;
      int m = mt + n;
#pragma unroll
      for (int p = 0; p < 8; ++p)
        af.u[p] = *(const unsigned*)(hsrc + m * HID + kk + kpat(p) + kh * 8);
#pragma unroll
      for (int q = 0; q < 4; ++q) {
        FragAB bf_;
        int rr = q * 16 + n;
#pragma unroll
        for (int p = 0; p < 8; ++p)
          bf_.u[p] = *(const unsigned*)(&Ws[rr * 512 + kk + kpat(p) + kh * 8]);
        acc[q] = __builtin_amdgcn_wmma_f32_16x16x32_bf16(
            false, af.v, false, bf_.v, (short)0, acc[q], false, false);
      }
    }

    // fused gates + state update + stores
    unsigned short* hdst = hbuf + (cur ^ 1) * (BATCH * HID);
#pragma unroll
    for (int r = 0; r < 8; ++r) {
      int m = mt + r + 8 * kh;
      float ig = sigmoidf(acc[0][r]);
      float fg = sigmoidf(acc[1][r]);
      float gg = tanhf(acc[2][r]);
      float og = sigmoidf(acc[3][r]);
      float cv = fg * c[r] + ig * gg;
      float hv = og * tanhf(cv);
      c[r] = cv;
      out[(size_t)t * BATCH * HID + m * HID + j0 + n] = hv;
      hdst[m * HID + j0 + n] = f2bf(hv);
      if (t == T_STEPS - 1) {
        hn[m * HID + j0 + n] = hv;
        cn[m * HID + j0 + n] = cv;
      }
    }

    // device-wide barrier (counter + generation)
    __threadfence();
    __syncthreads();
    if (tid == 0) {
      unsigned g0 = atomicAdd(&sync[1], 0u);
      unsigned arr = atomicAdd(&sync[0], 1u);
      if (arr == NWG_REC - 1) {
        atomicExch(&sync[0], 0u);
        __threadfence();
        atomicAdd(&sync[1], 1u);
      } else {
        volatile unsigned* vg = &sync[1];
        while (*vg == g0) __builtin_amdgcn_s_sleep(1);
      }
    }
    __syncthreads();
    __threadfence();
    cur ^= 1;
  }
}

// ---------------------------------------------------------------- launcher
extern "C" void kernel_launch(void* const* d_in, const int* in_sizes, int n_in,
                              void* d_out, int out_size, void* d_ws,
                              size_t ws_size, hipStream_t stream) {
  (void)in_sizes; (void)n_in; (void)out_size; (void)ws_size;
  const float* x     = (const float*)d_in[0];
  const float* w_ih0 = (const float*)d_in[1];
  const float* w_hh0 = (const float*)d_in[2];
  const float* b_ih0 = (const float*)d_in[3];
  const float* b_hh0 = (const float*)d_in[4];
  const float* w_ih1 = (const float*)d_in[5];
  const float* w_hh1 = (const float*)d_in[6];
  const float* b_ih1 = (const float*)d_in[7];
  const float* b_hh1 = (const float*)d_in[8];
  const float* h0    = (const float*)d_in[9];
  const float* c0    = (const float*)d_in[10];

  float* out1 = (float*)d_out;                                    // (T,B,H)
  float* hn   = out1 + (size_t)T_STEPS * BATCH * HID;             // (L,B,H)
  float* cn   = hn + 2 * BATCH * HID;                             // (L,B,H)

  char* ws = (char*)d_ws;
  size_t off = 0;
  auto take = [&](size_t bytes) {
    off = (off + 255) & ~(size_t)255;
    void* p = ws + off;
    off += bytes;
    return p;
  };
  float*          gi    = (float*)take((size_t)T_STEPS * BATCH * G4H * 4);
  float*          out0  = (float*)take((size_t)T_STEPS * BATCH * HID * 4);
  unsigned short* wih0b = (unsigned short*)take((size_t)G4H * IN_DIM * 2);
  unsigned short* whh0b = (unsigned short*)take((size_t)G4H * HID * 2);
  unsigned short* wih1b = (unsigned short*)take((size_t)G4H * HID * 2);
  unsigned short* whh1b = (unsigned short*)take((size_t)G4H * HID * 2);
  unsigned short* hbuf  = (unsigned short*)take(2 * (size_t)BATCH * HID * 2);
  unsigned*       sync  = (unsigned*)take(256);

  const int nW = G4H * 512;  // 1,048,576 elems per weight matrix
  cvt_bf16_kernel<<<nW / 256, 256, 0, stream>>>(w_ih0, wih0b, nW);
  cvt_bf16_kernel<<<nW / 256, 256, 0, stream>>>(w_hh0, whh0b, nW);
  cvt_bf16_kernel<<<nW / 256, 256, 0, stream>>>(w_ih1, wih1b, nW);
  cvt_bf16_kernel<<<nW / 256, 256, 0, stream>>>(w_hh1, whh1b, nW);

  dim3 ggrid(G4H / 32, (T_STEPS * BATCH) / 128);  // (64, 256)

  // layer 0
  gemm_gates_kernel<<<ggrid, 256, 0, stream>>>(x, wih0b, b_ih0, b_hh0, gi);
  init_state_kernel<<<64, 256, 0, stream>>>(h0, hbuf, sync);
  lstm_rec_kernel<<<NWG_REC, 64, 0, stream>>>(gi, whh0b, c0, hbuf, out0,
                                              hn, cn, sync);
  // layer 1
  gemm_gates_kernel<<<ggrid, 256, 0, stream>>>(out0, wih1b, b_ih1, b_hh1, gi);
  init_state_kernel<<<64, 256, 0, stream>>>(h0 + BATCH * HID, hbuf, sync);
  lstm_rec_kernel<<<NWG_REC, 64, 0, stream>>>(gi, whh1b, c0 + BATCH * HID,
                                              hbuf, out1, hn + BATCH * HID,
                                              cn + BATCH * HID, sync);
}
